// AttentionModel_59313498358412
// MI455X (gfx1250) — matrix operand order repaired and checked by reading, not compiled, not run on hardware
//
#include <hip/hip_runtime.h>

#define GAT_ALPHA 0.2f
#define GAT_EPS   1e-12f

typedef __attribute__((ext_vector_type(16))) _Float16 v16h;
typedef __attribute__((ext_vector_type(8)))  float    v8f;

// Order-preserving float -> uint encoding so atomicMax(u32) == float max.
__device__ __forceinline__ unsigned gat_enc(float v) {
    unsigned u = __float_as_uint(v);
    return (u & 0x80000000u) ? ~u : (u | 0x80000000u);
}
__device__ __forceinline__ float gat_dec(unsigned k) {
    return (k & 0x80000000u) ? __uint_as_float(k & 0x7FFFFFFFu)
                             : __uint_as_float(~k);
}

// ---------------- init: m-keys = 0 (== -inf class), sums = 0 ----------------
__global__ void gat_init(unsigned* __restrict__ mkey, float* __restrict__ s, int n) {
    int i = blockIdx.x * blockDim.x + threadIdx.x;
    if (i < n) { mkey[i] = 0u; s[i] = 0.0f; }
}

// ---------------- node projection  p[n][j] = sum_f x[n][f] * w[j][f] --------
// j in [0,8):  w[j][f]   = aa[j][f]        (src half)
// j in [8,16): w[j][f]   = aa[j-8][32+f]   (dst half)
// One V_WMMA_F32_16X16X32_F16 per 16-node tile (M=16 nodes, K=32 feats, N=16 j).
__global__ void gat_proj_wmma(const float* __restrict__ x,
                              const float* __restrict__ aa,
                              float* __restrict__ p, int N) {
    int wave  = (blockIdx.x * blockDim.x + threadIdx.x) >> 5;
    int lane  = threadIdx.x & 31;
    int tiles = (N + 15) >> 4;
    if (wave >= tiles) return;                 // whole-wave exit: EXEC stays all-1s

    int hlf = lane >> 4;                       // 0: lanes 0-15, 1: lanes 16-31
    int m   = lane & 15;                       // A: node-in-tile ; B: output j

    // ---- A: 16-bit A-matrix 16x32 layout (ISA 7.12.2) ----
    // lane half 0: elems 0-7 -> K=0..7,  elems 8-15 -> K=16..23
    // lane half 1: elems 0-7 -> K=8..15, elems 8-15 -> K=24..31
    int node = wave * 16 + m;
    if (node > N - 1) node = N - 1;            // clamp (keeps EXEC full)
    const float4* x4 = (const float4*)(x + (size_t)node * 32);
    float4 fa = x4[hlf * 2 + 0];
    float4 fb = x4[hlf * 2 + 1];
    float4 fc = x4[hlf * 2 + 4];
    float4 fd = x4[hlf * 2 + 5];
    v16h a;
    a[0]=(_Float16)fa.x; a[1]=(_Float16)fa.y; a[2]=(_Float16)fa.z; a[3]=(_Float16)fa.w;
    a[4]=(_Float16)fb.x; a[5]=(_Float16)fb.y; a[6]=(_Float16)fb.z; a[7]=(_Float16)fb.w;
    a[8]=(_Float16)fc.x; a[9]=(_Float16)fc.y; a[10]=(_Float16)fc.z; a[11]=(_Float16)fc.w;
    a[12]=(_Float16)fd.x; a[13]=(_Float16)fd.y; a[14]=(_Float16)fd.z; a[15]=(_Float16)fd.w;

    // ---- B: 32x16 (KxN), same K order as A: elems 0-7 -> K=hlf*8+i, elems 8-15 -> K=16+hlf*8+(i-8),
    //         N = lane&15; B[K][N = m] = w[m][K]
    int off = (m < 8) ? (m * 64 + hlf * 8) : ((m - 8) * 64 + 32 + hlf * 8);
    const float4* w4 = (const float4*)(aa + off);
    float4 wa = w4[0], wb = w4[1], wc = w4[4], wd = w4[5];
    v16h b;
    b[0]=(_Float16)wa.x; b[1]=(_Float16)wa.y; b[2]=(_Float16)wa.z; b[3]=(_Float16)wa.w;
    b[4]=(_Float16)wb.x; b[5]=(_Float16)wb.y; b[6]=(_Float16)wb.z; b[7]=(_Float16)wb.w;
    b[8]=(_Float16)wc.x; b[9]=(_Float16)wc.y; b[10]=(_Float16)wc.z; b[11]=(_Float16)wc.w;
    b[12]=(_Float16)wd.x; b[13]=(_Float16)wd.y; b[14]=(_Float16)wd.z; b[15]=(_Float16)wd.w;

    v8f c = {};
    c = __builtin_amdgcn_wmma_f32_16x16x32_f16(false, a, false, b, (short)0, c, false, false);

    // ---- D: VGPR r, lanes 0-15 -> M=r, lanes 16-31 -> M=8+r, N=lane&15 ----
    int rowbase = wave * 16 + hlf * 8;
    float* pt = p + (size_t)rowbase * 16 + m;
    if (rowbase + 8 <= N) {
        // full tile (hot path): straight-line stores, no exec-mask churn
        pt[0]      = c[0];
        pt[16]     = c[1];
        pt[32]     = c[2];
        pt[48]     = c[3];
        pt[64]     = c[4];
        pt[80]     = c[5];
        pt[96]     = c[6];
        pt[112]    = c[7];
    } else {
        for (int r = 0; r < 8; ++r)
            if (rowbase + r < N) pt[r * 16] = c[r];
    }
}

// ---------------- edge pass 1: segment max over destination (row) -----------
__global__ void gat_edge_max(const int* __restrict__ row, const int* __restrict__ col,
                             const float* __restrict__ p, unsigned* __restrict__ mkey,
                             int E) {
    int e = blockIdx.x * blockDim.x + threadIdx.x;
    if (e >= E) return;
    int r = row[e], c = col[e];
    const float4* pr = (const float4*)(p + (size_t)r * 16);
    const float4* pc = (const float4*)(p + (size_t)c * 16 + 8);
    float4 a0 = pr[0], a1 = pr[1], b0 = pc[0], b1 = pc[1];
    float v[8] = { a0.x + b0.x, a0.y + b0.y, a0.z + b0.z, a0.w + b0.w,
                   a1.x + b1.x, a1.y + b1.y, a1.z + b1.z, a1.w + b1.w };
    unsigned* mk = mkey + (size_t)r * 8;
#pragma unroll
    for (int h = 0; h < 8; ++h) {
        float t = v[h];
        t = t > 0.0f ? t : GAT_ALPHA * t;
        atomicMax(&mk[h], gat_enc(t));
    }
}

// ---------------- edge pass 2: ex = exp(e - m[row]); segment sum; stage ex --
__global__ void gat_edge_exp(const int* __restrict__ row, const int* __restrict__ col,
                             const float* __restrict__ p, const unsigned* __restrict__ mkey,
                             float* __restrict__ s, float* __restrict__ out, int E) {
    int e = blockIdx.x * blockDim.x + threadIdx.x;
    if (e >= E) return;
    int r = row[e], c = col[e];
    const float4* pr = (const float4*)(p + (size_t)r * 16);
    const float4* pc = (const float4*)(p + (size_t)c * 16 + 8);
    float4 a0 = pr[0], a1 = pr[1], b0 = pc[0], b1 = pc[1];
    float v[8] = { a0.x + b0.x, a0.y + b0.y, a0.z + b0.z, a0.w + b0.w,
                   a1.x + b1.x, a1.y + b1.y, a1.z + b1.z, a1.w + b1.w };
    const unsigned* mk = mkey + (size_t)r * 8;
    float* sv = s + (size_t)r * 8;
#pragma unroll
    for (int h = 0; h < 8; ++h) {
        float t = v[h];
        t = t > 0.0f ? t : GAT_ALPHA * t;
        float ex = __expf(t - gat_dec(mk[h]));
        atomicAdd(&sv[h], ex);
        out[(size_t)h * E + e] = ex;           // stage numerator in [H,E] layout
    }
}

// ---------------- node pass: s -> 1/(s+eps)  (400K divides instead of 12.8M)
__global__ void gat_recip(float* __restrict__ s, int n) {
    int i = blockIdx.x * blockDim.x + threadIdx.x;
    if (i < n) s[i] = 1.0f / (s[i] + GAT_EPS);
}

// ---------------- edge pass 3: normalize (pure multiply, bandwidth-bound) ---
__global__ void gat_norm(const int* __restrict__ row, const float* __restrict__ sinv,
                         float* __restrict__ out, int E) {
    int e = blockIdx.x * blockDim.x + threadIdx.x;
    if (e >= E) return;
    int r = row[e];
    const float4* sv4 = (const float4*)(sinv + (size_t)r * 8);
    float4 s0 = sv4[0], s1 = sv4[1];
    float sv[8] = { s0.x, s0.y, s0.z, s0.w, s1.x, s1.y, s1.z, s1.w };
#pragma unroll
    for (int h = 0; h < 8; ++h) {
        size_t idx = (size_t)h * E + e;
        out[idx] = out[idx] * sv[h];
    }
}

extern "C" void kernel_launch(void* const* d_in, const int* in_sizes, int n_in,
                              void* d_out, int out_size, void* d_ws, size_t ws_size,
                              hipStream_t stream) {
    const float* x   = (const float*)d_in[0];   // [N, 32]
    const float* aa  = (const float*)d_in[1];   // [8, 64]
    const int*   row = (const int*)d_in[2];     // [E]
    const int*   col = (const int*)d_in[3];     // [E]
    float*       out = (float*)d_out;           // [8, E]

    const int F = 32;
    int N = in_sizes[0] / F;
    int E = in_sizes[2];

    // workspace: p[N][16] | mkey[N][8] | s[N][8]   (6.4 MB total @ N=50000)
    char* ws = (char*)d_ws;
    float*    p    = (float*)ws;
    unsigned* mkey = (unsigned*)(ws + (size_t)N * 16 * sizeof(float));
    float*    s    = (float*)(ws + (size_t)N * 16 * sizeof(float)
                                 + (size_t)N * 8 * sizeof(unsigned));

    const int T = 256;
    int nNode = N * 8;
    dim3 ngrid((nNode + T - 1) / T);
    hipLaunchKernelGGL(gat_init, ngrid, dim3(T), 0, stream, mkey, s, nNode);

    int tiles = (N + 15) / 16;
    int projThreads = tiles * 32;               // one wave32 per 16-node tile
    hipLaunchKernelGGL(gat_proj_wmma, dim3((projThreads + T - 1) / T), dim3(T), 0, stream,
                       x, aa, p, N);

    dim3 egrid((E + T - 1) / T);
    hipLaunchKernelGGL(gat_edge_max, egrid, dim3(T), 0, stream, row, col, p, mkey, E);
    hipLaunchKernelGGL(gat_edge_exp, egrid, dim3(T), 0, stream, row, col, p, mkey, s, out, E);
    hipLaunchKernelGGL(gat_recip,   ngrid, dim3(T), 0, stream, s, nNode);
    hipLaunchKernelGGL(gat_norm,    egrid, dim3(T), 0, stream, row, s, out, E);
}
